// SelfAttentionV2_24120536334832
// MI455X (gfx1250) — matrix operand order repaired and checked by reading, not compiled, not run on hardware
//
#include <hip/hip_runtime.h>
#include <hip/hip_bf16.h>
#include <stdint.h>

typedef __attribute__((ext_vector_type(16))) __bf16 v16bf;
typedef __attribute__((ext_vector_type(8)))  __bf16 v8bf;
typedef __attribute__((ext_vector_type(8)))  float  v8f;

#define SEQ  4096
#define DIN  768
#define DOUT 768

// LDS B-panel: 128 columns (n) x KC (k) bf16, double buffered.
// Row pitch 72 elements = 144B -> 16B-aligned lane reads, conflict-free
// bank quads (36 dwords/row, gcd(36,64)=4 -> 16-row period covers 64 banks).
#define KC   64
#define LDB  72
#define NT   8          // 16x16 output tiles per wave (wave tile = 16 x 128)

__device__ __forceinline__ __bf16 to_bf16(float f) { return (__bf16)f; }

// ---------------------------------------------------------------------------
// Elementwise f32 -> bf16
// ---------------------------------------------------------------------------
__global__ void cvt_f32_to_bf16(const float* __restrict__ in,
                                __bf16* __restrict__ out, int n) {
    int i = blockIdx.x * blockDim.x + threadIdx.x;
    if (i < n) out[i] = to_bf16(in[i]);
}

// ---------------------------------------------------------------------------
// Transposing convert: in [R][C] f32 (row-major) -> out [C][R] bf16
// ---------------------------------------------------------------------------
__global__ void cvt_transpose_bf16(const float* __restrict__ in,
                                   __bf16* __restrict__ out, int R, int C) {
    int c = blockIdx.x * blockDim.x + threadIdx.x;
    int r = blockIdx.y * blockDim.y + threadIdx.y;
    if (r < R && c < C) out[(size_t)c * R + r] = to_bf16(in[(size_t)r * C + c]);
}

// ---------------------------------------------------------------------------
// bf16 WMMA GEMM, async-LDS-staged B panel, 16x128 wave tile:
//   C[M][N] = scale * (A[M][K] @ Bt[N][K]^T)
// A  : row-major bf16 [M][K]  (streamed from global per wave)
// Bt : row-major bf16 [N][K]  (128-row panel staged to LDS per block via
//      GLOBAL_LOAD_ASYNC_TO_LDS_B128, double buffered, ASYNCcnt-tracked)
// STORE_MODE: 0 = f32 [M][N], 1 = bf16 [M][N], 2 = bf16 transposed [N][M]
// Block = 256 threads = 8 waves covering 128(M) x 128(N).
// Requires K % KC == 0, M % 128 == 0, N % 128 == 0.
// ---------------------------------------------------------------------------
template <int STORE_MODE>
__global__ __launch_bounds__(256)
void gemm_bf16_wmma(const __bf16* __restrict__ A,
                    const __bf16* __restrict__ Bt,
                    void* __restrict__ out,
                    int M, int N, int K, float scale) {
    __shared__ __bf16 ldsB[2][128 * LDB];

    const int lane = threadIdx.x & 31;
    const int wave = threadIdx.x >> 5;
    const int m0   = blockIdx.y * 128 + wave * 16;
    const int n0   = blockIdx.x * 128;
    const int mrow = lane & 15;   // A row / C column within tile
    const int half = lane >> 4;

    const uint32_t ldsBase = (uint32_t)(uintptr_t)&ldsB[0][0];

    // Stage the 128 x KC bf16 panel for `chunk` into LDS buffer `buf`.
    // 128 rows * 8 b128 transfers = 1024; 256 threads -> 4 each.
    auto issue_copy = [&](int chunk, int buf) {
        const __bf16* gbase = Bt + (size_t)n0 * K + (size_t)chunk * KC;
#pragma unroll
        for (int i = 0; i < 4; ++i) {
            const int e = (int)threadIdx.x + i * 256;
            const int r = e >> 3;        // panel row 0..127
            const int c = (e & 7) * 8;   // k element offset (16B granules)
            const uint32_t loff = ldsBase +
                (uint32_t)(buf * (128 * LDB) + r * LDB + c) * 2u;
            const uint64_t ga = (uint64_t)(uintptr_t)(gbase + (size_t)r * K + c);
            asm volatile("global_load_async_to_lds_b128 %0, %1, off"
                         :: "v"(loff), "v"(ga) : "memory");
        }
    };

    // Load one 32x16 B fragment from the LDS panel.
    auto load_bfrag = [&](const __bf16* bl, int tile, int kc) -> v16bf {
        const __bf16* bp = bl + (tile * 16 + mrow) * LDB + kc + half * 8;
        v8bf blo = *(const v8bf*)(bp);
        v8bf bhi = *(const v8bf*)(bp + 16);
        return __builtin_shufflevector(blo, bhi,
                   0, 1, 2, 3, 4, 5, 6, 7, 8, 9, 10, 11, 12, 13, 14, 15);
    };

    v8f acc[NT] = {};
    const __bf16* __restrict__ arow = A + (size_t)(m0 + mrow) * K;
    const int nch = K / KC;

    issue_copy(0, 0);

    for (int ch = 0; ch < nch; ++ch) {
        asm volatile("s_wait_asynccnt 0x0" ::: "memory");
        __syncthreads();
        if (ch + 1 < nch) issue_copy(ch + 1, (ch + 1) & 1);

        const __bf16* bl = &ldsB[ch & 1][0];
        const int kg = ch * KC;
#pragma unroll
        for (int kc = 0; kc < KC; kc += 32) {
            // A fragment (16x32): lanes 0-15 hold K 0-7 & 16-23 of row m,
            // lanes 16-31 hold K 8-15 & 24-31.
            v8bf alo = *(const v8bf*)(arow + kg + kc +      half * 8);
            v8bf ahi = *(const v8bf*)(arow + kg + kc + 16 + half * 8);
            v16bf a  = __builtin_shufflevector(alo, ahi,
                          0, 1, 2, 3, 4, 5, 6, 7, 8, 9, 10, 11, 12, 13, 14, 15);
            // Two groups of 4 tiles. The empty asm forces the 4 fragments to
            // be simultaneously live in distinct register tuples, so the ds
            // loads all issue before the WMMAs and waits become partial.
#pragma unroll
            for (int g = 0; g < NT / 4; ++g) {
                v16bf vb0 = load_bfrag(bl, g * 4 + 0, kc);
                v16bf vb1 = load_bfrag(bl, g * 4 + 1, kc);
                v16bf vb2 = load_bfrag(bl, g * 4 + 2, kc);
                v16bf vb3 = load_bfrag(bl, g * 4 + 3, kc);
                asm volatile("" : "+v"(vb0), "+v"(vb1), "+v"(vb2), "+v"(vb3));
                acc[g * 4 + 0] = __builtin_amdgcn_wmma_f32_16x16x32_bf16(
                    false, a, false, vb0, (short)0, acc[g * 4 + 0], false, false);
                acc[g * 4 + 1] = __builtin_amdgcn_wmma_f32_16x16x32_bf16(
                    false, a, false, vb1, (short)0, acc[g * 4 + 1], false, false);
                acc[g * 4 + 2] = __builtin_amdgcn_wmma_f32_16x16x32_bf16(
                    false, a, false, vb2, (short)0, acc[g * 4 + 2], false, false);
                acc[g * 4 + 3] = __builtin_amdgcn_wmma_f32_16x16x32_bf16(
                    false, a, false, vb3, (short)0, acc[g * 4 + 3], false, false);
            }
        }
    }

    // C/D layout: VGPR r, lanes 0-15 -> (M=r, N=lane); lanes 16-31 -> (M=r+8).
#pragma unroll
    for (int t = 0; t < NT; ++t) {
        const int n = n0 + t * 16 + mrow;
#pragma unroll
        for (int r = 0; r < 8; ++r) {
            const int m = m0 + r + half * 8;
            const float v = acc[t][r] * scale;
            if (STORE_MODE == 0) {
                ((float*)out)[(size_t)m * N + n] = v;
            } else if (STORE_MODE == 1) {
                ((__bf16*)out)[(size_t)m * N + n] = to_bf16(v);
            } else {
                ((__bf16*)out)[(size_t)n * M + m] = to_bf16(v);
            }
        }
    }
}

// ---------------------------------------------------------------------------
// Row softmax: one 256-thread block per row of S [rows][4096] f32 -> P bf16.
// ---------------------------------------------------------------------------
__global__ __launch_bounds__(256)
void softmax_row(const float* __restrict__ S, __bf16* __restrict__ P, int n) {
    __shared__ float red[256];
    const int row = blockIdx.x;
    const int tid = threadIdx.x;
    const float* __restrict__ srow = S + (size_t)row * n;

    float vals[16];
    float lmax = -3.0e38f;
#pragma unroll
    for (int i = 0; i < 16; ++i) {
        vals[i] = srow[tid + i * 256];
        lmax = fmaxf(lmax, vals[i]);
    }
    red[tid] = lmax;
    __syncthreads();
    for (int s = 128; s > 0; s >>= 1) {
        if (tid < s) red[tid] = fmaxf(red[tid], red[tid + s]);
        __syncthreads();
    }
    const float rmax = red[0];
    __syncthreads();

    float lsum = 0.0f;
#pragma unroll
    for (int i = 0; i < 16; ++i) {
        vals[i] = __expf(vals[i] - rmax);
        lsum += vals[i];
    }
    red[tid] = lsum;
    __syncthreads();
    for (int s = 128; s > 0; s >>= 1) {
        if (tid < s) red[tid] += red[tid + s];
        __syncthreads();
    }
    const float inv = 1.0f / red[0];

    __bf16* __restrict__ prow = P + (size_t)row * n;
#pragma unroll
    for (int i = 0; i < 16; ++i) prow[tid + i * 256] = to_bf16(vals[i] * inv);
}

// ---------------------------------------------------------------------------
// Launch
// ---------------------------------------------------------------------------
extern "C" void kernel_launch(void* const* d_in, const int* in_sizes, int n_in,
                              void* d_out, int out_size, void* d_ws, size_t ws_size,
                              hipStream_t stream) {
    (void)in_sizes; (void)n_in; (void)out_size; (void)ws_size;

    const float* x  = (const float*)d_in[0];   // [SEQ][DIN]
    const float* Wq = (const float*)d_in[1];   // [DIN][DOUT]
    const float* Wk = (const float*)d_in[2];
    const float* Wv = (const float*)d_in[3];
    float* out = (float*)d_out;                // [SEQ][DOUT] f32

    char* ws = (char*)d_ws;
    size_t off = 0;
    auto carve = [&](size_t bytes) -> char* {
        char* p = ws + off;
        off += (bytes + 255) & ~(size_t)255;
        return p;
    };
    __bf16* Xb  = (__bf16*)carve((size_t)SEQ * DIN * 2);   // x bf16
    __bf16* Wqt = (__bf16*)carve((size_t)DIN * DOUT * 2);  // Wq^T bf16
    __bf16* Wkt = (__bf16*)carve((size_t)DIN * DOUT * 2);
    __bf16* Wvt = (__bf16*)carve((size_t)DIN * DOUT * 2);
    __bf16* Qb  = (__bf16*)carve((size_t)SEQ * DOUT * 2);  // Q bf16
    __bf16* Kb  = (__bf16*)carve((size_t)SEQ * DOUT * 2);  // K bf16
    __bf16* Vt  = (__bf16*)carve((size_t)DOUT * SEQ * 2);  // V^T bf16
    float*  Sc  = (float*) carve((size_t)SEQ * SEQ * 4);   // scores f32 (L2)
    __bf16* Pb  = (__bf16*)carve((size_t)SEQ * SEQ * 2);   // probs bf16

    // Phase 0: downconvert inputs (weights transposed)
    {
        const int n = SEQ * DIN;
        cvt_f32_to_bf16<<<(n + 255) / 256, 256, 0, stream>>>(x, Xb, n);
        dim3 tb(16, 16), tg(DOUT / 16, DIN / 16);
        cvt_transpose_bf16<<<tg, tb, 0, stream>>>(Wq, Wqt, DIN, DOUT);
        cvt_transpose_bf16<<<tg, tb, 0, stream>>>(Wk, Wkt, DIN, DOUT);
        cvt_transpose_bf16<<<tg, tb, 0, stream>>>(Wv, Wvt, DIN, DOUT);
    }

    // Phase 1: Q, K (bf16), V^T (bf16 transposed store)
    {
        dim3 grid(DOUT / 128, SEQ / 128);
        gemm_bf16_wmma<1><<<grid, 256, 0, stream>>>(Xb, Wqt, Qb, SEQ, DOUT, DIN, 1.0f);
        gemm_bf16_wmma<1><<<grid, 256, 0, stream>>>(Xb, Wkt, Kb, SEQ, DOUT, DIN, 1.0f);
        gemm_bf16_wmma<2><<<grid, 256, 0, stream>>>(Xb, Wvt, Vt, SEQ, DOUT, DIN, 1.0f);
    }

    // Phase 2: scores = (Q @ K^T) / sqrt(DOUT)
    {
        dim3 grid(SEQ / 128, SEQ / 128);
        const float scale = 1.0f / sqrtf((float)DOUT);
        gemm_bf16_wmma<0><<<grid, 256, 0, stream>>>(Qb, Kb, Sc, SEQ, SEQ, DOUT, scale);
    }

    // Phase 3: row softmax -> P (bf16)
    softmax_row<<<SEQ, 256, 0, stream>>>(Sc, Pb, SEQ);

    // Phase 4: out = P @ V
    {
        dim3 grid(DOUT / 128, SEQ / 128);
        gemm_bf16_wmma<0><<<grid, 256, 0, stream>>>(Pb, Vt, out, SEQ, DOUT, SEQ, 1.0f);
    }
}
